// Protonet_29832842838476
// MI455X (gfx1250) — matrix operand order repaired and checked by reading, not compiled, not run on hardware
//
#include <hip/hip_runtime.h>
#include <hip/hip_bf16.h>

// ---------------------------------------------------------------------------
// Protonet forward on MI455X (gfx1250)
//   encoder GEMMs: bf16 v_wmma_f32_16x16x32_bf16, double-buffered LDS tiles
//   staged with async global->LDS loads (ASYNCcnt) when available.
// ---------------------------------------------------------------------------

typedef __attribute__((ext_vector_type(16))) __bf16 v16bf;
typedef __attribute__((ext_vector_type(8)))  __bf16 v8bf;
typedef __attribute__((ext_vector_type(8)))  float  v8f;
typedef __attribute__((ext_vector_type(4)))  int    v4i;

// address-space-qualified int4 for the async global->LDS builtin
typedef __attribute__((address_space(1))) v4i glob_v4i;   // global
typedef __attribute__((address_space(3))) v4i lds_v4i;    // LDS

#define BM 128
#define BN 128
#define BK 64          // two 16x16x32 k-substeps per LDS tile

#if defined(__AMDGCN__) && defined(__has_builtin)
#if __has_builtin(__builtin_amdgcn_global_load_async_to_lds_b128)
#define HAVE_ASYNC_LDS 1
#endif
#endif
#ifndef HAVE_ASYNC_LDS
#define HAVE_ASYNC_LDS 0
#endif

__device__ __forceinline__ __bf16 f2bf(float f) { return (__bf16)f; }

// 16B global -> LDS copy; async DMA path on CDNA5, sync fallback otherwise.
__device__ __forceinline__ void copy16_g2l(__bf16* lds, const __bf16* g) {
#if HAVE_ASYNC_LDS
    __builtin_amdgcn_global_load_async_to_lds_b128(
        (glob_v4i*)(g), (lds_v4i*)(lds), 0, 0);
#else
    *(uint4*)lds = *(const uint4*)g;
#endif
}

__device__ __forceinline__ void wait_async_done() {
#if HAVE_ASYNC_LDS
#if __has_builtin(__builtin_amdgcn_s_wait_asynccnt)
    __builtin_amdgcn_s_wait_asynccnt(0);
#else
    asm volatile("s_wait_asynccnt 0" ::: "memory");
#endif
#endif
}

// A fragment (16x32 bf16, MxK): lane l holds row M=l%16.
// lanes 0-15: elems 0-7 -> K=kb..kb+7, elems 8-15 -> K=kb+16..kb+23 (kb=0)
// lanes 16-31: same with kb=8.   LDS tile stride = BK halves.
__device__ __forceinline__ v16bf load_frag_A(const __bf16* tile, int lane) {
    int m  = lane & 15;
    int kb = (lane >> 4) * 8;
    const v8bf* p0 = (const v8bf*)(tile + m * BK + kb);
    const v8bf* p1 = (const v8bf*)(tile + m * BK + kb + 16);
    return __builtin_shufflevector(*p0, *p1,
        0,1,2,3,4,5,6,7,8,9,10,11,12,13,14,15);
}

// B fragment (32x16 bf16, KxN), tile stored [N][K] K-contiguous:
// lane l holds col N=l%16; elems 0-7 -> K=kb..kb+7, elems 8-15 -> K=kb+16..kb+23, kb=8*(l>=16)
__device__ __forceinline__ v16bf load_frag_B(const __bf16* tile, int lane) {
    int n  = lane & 15;
    int kb = (lane >> 4) * 8;
    const v8bf* p0 = (const v8bf*)(tile + n * BK + kb);
    const v8bf* p1 = (const v8bf*)(tile + n * BK + kb + 16);
    return __builtin_shufflevector(*p0, *p1,
        0,1,2,3,4,5,6,7,8,9,10,11,12,13,14,15);
}

// C = relu?(A @ Bt^T + bias); A: MxK bf16 row-major, Bt: NxK bf16 row-major.
template<bool RELU, bool OUT_BF16>
__global__ __launch_bounds__(256)
void wmma_gemm_kernel(const __bf16* __restrict__ A,
                      const __bf16* __restrict__ Bt,
                      const float* __restrict__ bias,
                      void* __restrict__ Cout,
                      int M, int N, int K) {
    __shared__ __attribute__((aligned(16))) __bf16 As[2][BM * BK];
    __shared__ __attribute__((aligned(16))) __bf16 Bs[2][BN * BK];

    const int tid   = threadIdx.x;
    const int lane  = tid & 31;
    const int wid   = tid >> 5;          // 0..7
    const int wm    = wid >> 2;          // 0..1  (64 rows each)
    const int wn    = wid & 3;           // 0..3  (32 cols each)
    const long mbase = (long)blockIdx.y * BM;
    const long nbase = (long)blockIdx.x * BN;

    v8f acc[4][2] = {};

    // staging: BM*BK/8 = 1024 16B-chunks per tile, 4 per thread; 8 halves/chunk
    auto stage = [&](int buf, int k0) {
        #pragma unroll
        for (int i = 0; i < 4; ++i) {
            int c   = tid + i * 256;
            int row = c >> 3;            // 0..127
            int off = (c & 7) * 8;       // 0..56 halves
            copy16_g2l(&As[buf][row * BK + off],
                       A + (mbase + row) * (long)K + k0 + off);
            copy16_g2l(&Bs[buf][row * BK + off],
                       Bt + (nbase + row) * (long)K + k0 + off);
        }
    };

    const int nk = K / BK;
    stage(0, 0);
    wait_async_done();
    __syncthreads();

    for (int kt = 0; kt < nk; ++kt) {
        const int cur = kt & 1;
        if (kt + 1 < nk) stage(1 - cur, (kt + 1) * BK);   // prefetch next tile

        #pragma unroll
        for (int ks = 0; ks < 2; ++ks) {                  // two 16x16x32 steps
            const __bf16* Ab = &As[cur][0] + ks * 32;
            const __bf16* Bb = &Bs[cur][0] + ks * 32;

            v16bf bfrag[2];
            #pragma unroll
            for (int nt = 0; nt < 2; ++nt)
                bfrag[nt] = load_frag_B(Bb + (wn * 32 + nt * 16) * BK, lane);

            #pragma unroll
            for (int mt = 0; mt < 4; ++mt) {
                v16bf afrag = load_frag_A(Ab + (wm * 64 + mt * 16) * BK, lane);
                #pragma unroll
                for (int nt = 0; nt < 2; ++nt) {
                    acc[mt][nt] = __builtin_amdgcn_wmma_f32_16x16x32_bf16(
                        false, afrag, false, bfrag[nt],
                        (short)0, acc[mt][nt], false, false);
                }
            }
        }
        wait_async_done();     // prefetched tile landed
        __syncthreads();       // everyone done reading cur / writing next
    }

    // ---- epilogue: bias (+ReLU) and store ----
    #pragma unroll
    for (int nt = 0; nt < 2; ++nt) {
        long col = nbase + wn * 32 + nt * 16 + (lane & 15);
        float bv = bias[col];
        #pragma unroll
        for (int mt = 0; mt < 4; ++mt) {
            long row0 = mbase + wm * 64 + mt * 16 + 8 * (lane >> 4);
            #pragma unroll
            for (int v = 0; v < 8; ++v) {
                float val = acc[mt][nt][v] + bv;
                if (RELU) val = val > 0.0f ? val : 0.0f;
                long idx = (row0 + v) * (long)N + col;
                if (OUT_BF16) ((__bf16*)Cout)[idx] = f2bf(val);
                else          ((float*)Cout)[idx]  = val;
            }
        }
    }
}

// concat(x_support, x_query) f32 -> bf16, one pass (keeps GEMM staging pure bf16)
__global__ void convert_x_kernel(const float* __restrict__ xs,
                                 const float* __restrict__ xq,
                                 __bf16* __restrict__ X, long splitElems) {
    long i = ((long)blockIdx.x * blockDim.x + threadIdx.x) * 8;
    const float* src = (i < splitElems) ? xs + i : xq + (i - splitElems);
    float4 f0 = ((const float4*)src)[0];
    float4 f1 = ((const float4*)src)[1];
    v8bf o;
    o[0] = f2bf(f0.x); o[1] = f2bf(f0.y); o[2] = f2bf(f0.z); o[3] = f2bf(f0.w);
    o[4] = f2bf(f1.x); o[5] = f2bf(f1.y); o[6] = f2bf(f1.z); o[7] = f2bf(f1.w);
    *(v8bf*)(X + i) = o;
}

// Transpose + convert: W (KxN f32) -> WT (NxK bf16)
__global__ void transpose_f32_to_bf16(const float* __restrict__ W,
                                      __bf16* __restrict__ WT,
                                      int K, int N) {
    __shared__ float tile[32][33];
    int nb = blockIdx.x * 32, kb = blockIdx.y * 32;
    int tx = threadIdx.x, ty = threadIdx.y;     // 32 x 8
    #pragma unroll
    for (int i = 0; i < 32; i += 8)
        tile[ty + i][tx] = W[(long)(kb + ty + i) * N + nb + tx];
    __syncthreads();
    #pragma unroll
    for (int i = 0; i < 32; i += 8)
        WT[(long)(nb + ty + i) * K + kb + tx] = f2bf(tile[tx][ty + i]);
}

// prototypes: mean over 32 support embeddings per class
__global__ void proto_mean_kernel(const float* __restrict__ Z,
                                  float* __restrict__ P) {
    int d = blockIdx.x * blockDim.x + threadIdx.x;  // 0..1023
    int c = blockIdx.y;                             // 0..127
    float s = 0.0f;
    #pragma unroll 4
    for (int j = 0; j < 32; ++j)
        s += Z[(long)(c * 32 + j) * 1024 + d];
    P[(long)c * 1024 + d] = s * (1.0f / 32.0f);
}

__global__ void zero_out_kernel(float* out) {
    if (threadIdx.x < 2) out[threadIdx.x] = 0.0f;
}

// per-query: dists to 128 prototypes, log-softmax, loss & acc accumulation
__global__ __launch_bounds__(128)
void loss_acc_kernel(const float* __restrict__ Zq,   // 4096 x 1024
                     const float* __restrict__ P,    // 128 x 1024
                     const int* __restrict__ y,      // 4096
                     float* __restrict__ out) {      // [loss, acc]
    __shared__ float qs[1024];
    __shared__ float logits[128];
    __shared__ float red[128];
    __shared__ int   redi[128];

    const int q = blockIdx.x;
    const int t = threadIdx.x;     // 0..127 = class id

    for (int j = t; j < 1024; j += 128) qs[j] = Zq[(long)q * 1024 + j];
    __syncthreads();

    const float* p = P + (long)t * 1024;
    float d = 0.0f;
    #pragma unroll 4
    for (int j = 0; j < 1024; ++j) {
        float df = qs[j] - p[j];
        d += df * df;
    }
    float lg = -d;
    logits[t] = lg;
    red[t] = lg; redi[t] = t;
    __syncthreads();

    // max + first-index argmax (matches jnp.argmax tie-break)
    for (int s = 64; s > 0; s >>= 1) {
        if (t < s) {
            float a = red[t], b = red[t + s];
            int ia = redi[t], ib = redi[t + s];
            if (b > a || (b == a && ib < ia)) { red[t] = b; redi[t] = ib; }
        }
        __syncthreads();
    }
    float mx = red[0];
    int amax = redi[0];
    __syncthreads();

    red[t] = expf(lg - mx);
    __syncthreads();
    for (int s = 64; s > 0; s >>= 1) {
        if (t < s) red[t] += red[t + s];
        __syncthreads();
    }

    if (t == 0) {
        int lbl = y[q];
        float logp = logits[lbl] - mx - logf(red[0]);
        atomicAdd(&out[0], -logp * (1.0f / 4096.0f));
        atomicAdd(&out[1], (amax == lbl) ? (1.0f / 4096.0f) : 0.0f);
    }
}

// ---------------------------------------------------------------------------

extern "C" void kernel_launch(void* const* d_in, const int* in_sizes, int n_in,
                              void* d_out, int out_size, void* d_ws, size_t ws_size,
                              hipStream_t stream) {
    (void)in_sizes; (void)n_in; (void)out_size; (void)ws_size;

    const int N_CLASS = 128, N_SUPPORT = 32, N_QUERY = 32;
    const int IN_DIM = 2048, HIDDEN = 4096, Z_DIM = 1024;
    const int M = N_CLASS * (N_SUPPORT + N_QUERY);         // 8192
    const long ROW_SPLIT = (long)N_CLASS * N_SUPPORT;      // 4096

    const float* x_support = (const float*)d_in[0];
    const float* x_query   = (const float*)d_in[1];
    const int*   y_query   = (const int*)d_in[2];
    const float* W1        = (const float*)d_in[3];
    const float* b1        = (const float*)d_in[4];
    const float* W2        = (const float*)d_in[5];
    const float* b2        = (const float*)d_in[6];
    float*       out       = (float*)d_out;

    // workspace layout (bytes)
    char* ws = (char*)d_ws;
    __bf16* Xbf = (__bf16*)(ws);                           // 8192 x 2048 (32MB)
    __bf16* W1T = (__bf16*)(ws + 33554432);                // 4096 x 2048 (16MB)
    __bf16* W2T = (__bf16*)(ws + 33554432 + 16777216);     // 1024 x 4096 (8MB)
    __bf16* H   = (__bf16*)(ws + 58720256);                // 8192 x 4096 (64MB)
    float*  Z   = (float*)(ws + 58720256 + 67108864);      // 8192 x 1024 (32MB)
    float*  P   = (float*)(ws + 58720256 + 67108864 + 33554432); // 128 x 1024

    // 1) input + weight conversion to bf16 (weights transposed to [N][K])
    convert_x_kernel<<<(M * IN_DIM / 8) / 256, 256, 0, stream>>>(
        x_support, x_query, Xbf, ROW_SPLIT * IN_DIM);
    transpose_f32_to_bf16<<<dim3(HIDDEN / 32, IN_DIM / 32), dim3(32, 8), 0, stream>>>(
        W1, W1T, IN_DIM, HIDDEN);
    transpose_f32_to_bf16<<<dim3(Z_DIM / 32, HIDDEN / 32), dim3(32, 8), 0, stream>>>(
        W2, W2T, HIDDEN, Z_DIM);

    // 2) GEMM1: H = relu(X @ W1 + b1)
    wmma_gemm_kernel<true, true><<<dim3(HIDDEN / BN, M / BM), 256, 0, stream>>>(
        Xbf, W1T, b1, H, M, HIDDEN, IN_DIM);

    // 3) GEMM2: Z = H @ W2 + b2
    wmma_gemm_kernel<false, false><<<dim3(Z_DIM / BN, M / BM), 256, 0, stream>>>(
        H, W2T, b2, Z, M, Z_DIM, HIDDEN);

    // 4) prototypes
    proto_mean_kernel<<<dim3(Z_DIM / 256, N_CLASS), 256, 0, stream>>>(Z, P);

    // 5) loss + accuracy
    zero_out_kernel<<<1, 32, 0, stream>>>(out);
    loss_acc_kernel<<<N_CLASS * N_QUERY, 128, 0, stream>>>(
        Z + ROW_SPLIT * Z_DIM, P, y_query, out);
}
